// SparseLinear_4131758539277
// MI455X (gfx1250) — matrix operand order repaired and checked by reading, not compiled, not run on hardware
//
#include <hip/hip_runtime.h>

typedef __attribute__((ext_vector_type(16))) _Float16 v16h;
typedef __attribute__((ext_vector_type(4)))  _Float16 v4h;
typedef __attribute__((ext_vector_type(8)))  float    v8f;

#define BATCH   4096
#define SEQ_L   200
#define NCH     128
#define KCHUNK  32
#define NCHUNKS 7        // ceil(200/32)
#define LDSROW  132      // halves per staged row: 128 + 4 pad -> conflict-free column reads

__global__ __launch_bounds__(32)
void embbag_wmma_kernel(const int*   __restrict__ x,
                        const float* __restrict__ w,
                        const float* __restrict__ table,
                        float*       __restrict__ out)
{
    __shared__ _Float16 ldsE[KCHUNK * LDSROW];   // staged f16 chunk: 32 rows x 128 ch (row-major)
    __shared__ float    ldsW[KCHUNK];            // staged weights for this chunk

    const int b    = blockIdx.x;      // one sample per wave/block
    const int lane = threadIdx.x;     // 0..31 (wave32)

    // 8 accumulators: 16x16 f32 tiles covering the 128 output channels
    v8f acc[8];
    const v8f vzero = {0.f,0.f,0.f,0.f,0.f,0.f,0.f,0.f};
#pragma unroll
    for (int i = 0; i < 8; ++i) acc[i] = vzero;

    const int baseA = (lane < 16) ? 0 : 8;    // table-order K base (ISA 7.12.2); unused, A uses koffB
    const int koffB = (lane < 16) ? 0 : 16;   // B-matrix K offset for this lane half
    const int ncol  = lane & 15;              // N (channel within 16-wide block)

    for (int ch = 0; ch < NCHUNKS; ++ch) {
        const int k0 = ch * KCHUNK;

        // ---- branchless index/weight fetch: lane <-> chunk row, coalesced ----
        const int l  = k0 + lane;
        const int lc = (l < SEQ_L) ? l : (SEQ_L - 1);   // clamp: loads stay valid
        const int   vIdx = x[b * SEQ_L + lc];           // one global_load_b32
        float       wv   = w[b * SEQ_L + lc];           // one global_load_b32
        wv = (l < SEQ_L) ? wv : 0.f;                    // padded rows weighted by 0
        ldsW[lane] = wv;

        // ---- stage: each lane streams ITS OWN 512B row; 16 independent b128
        //      loads in flight per group -> deep MLP, no scalar-load serialization ----
        const float4* rowp =
            reinterpret_cast<const float4*>(table + (size_t)(unsigned)vIdx * NCH);
#pragma unroll
        for (int g = 0; g < 2; ++g) {
            float4 rv[16];
#pragma unroll
            for (int j = 0; j < 16; ++j)            // 16 back-to-back global_load_b128
                rv[j] = rowp[g * 16 + j];
#pragma unroll
            for (int j = 0; j < 16; ++j) {          // f32 -> f16, 8B ds_store, conflict-free
                v4h h4;
                h4.x = (_Float16)rv[j].x; h4.y = (_Float16)rv[j].y;
                h4.z = (_Float16)rv[j].z; h4.w = (_Float16)rv[j].w;
                *reinterpret_cast<v4h*>(&ldsE[lane * LDSROW + (g * 16 + j) * 4]) = h4;
            }
        }
        // Same-wave DS ops are in-order: staged data visible to DS reads below.

        // ---- build A fragment: weight chunk broadcast to all 16 M rows ----
        union { v16h v; _Float16 h[16]; } A;
#pragma unroll
        for (int j = 0; j < 8; ++j) {
            const int kk = koffB + 2 * j;
            A.h[2 * j]     = (_Float16)ldsW[kk];
            A.h[2 * j + 1] = (_Float16)ldsW[kk + 1];
        }

        // ---- 8 column blocks of 16 channels: build B (32x16) and WMMA ----
#pragma unroll
        for (int cb = 0; cb < 8; ++cb) {
            union { v16h v; _Float16 h[16]; } Bf;
            const int c = cb * 16 + ncol;
#pragma unroll
            for (int j = 0; j < 16; ++j) {            // element j <-> K = koffB + j
                Bf.h[j] = ldsE[(koffB + j) * LDSROW + c];
            }
            acc[cb] = __builtin_amdgcn_wmma_f32_16x16x32_f16(
                /*neg_a=*/false, A.v, /*neg_b=*/false, Bf.v,
                /*c_mod=*/(short)0, acc[cb],
                /*reuse_a=*/false, /*reuse_b=*/false);
        }
    }

    // ---- epilogue: all 16 D rows identical -> lanes 0..15 of element 0 hold the channels ----
    if (lane < 16) {
#pragma unroll
        for (int cb = 0; cb < 8; ++cb) {
            out[b * NCH + cb * 16 + lane] = acc[cb][0];
        }
    }
}

extern "C" void kernel_launch(void* const* d_in, const int* in_sizes, int n_in,
                              void* d_out, int out_size, void* d_ws, size_t ws_size,
                              hipStream_t stream) {
    (void)in_sizes; (void)n_in; (void)out_size; (void)d_ws; (void)ws_size;
    const int*   x     = (const int*)  d_in[0];   // (4096, 200) int32
    const float* w     = (const float*)d_in[1];   // (4096, 200) f32
    const float* table = (const float*)d_in[2];   // (1e6, 128) f32
    float*       out   = (float*)d_out;           // (4096, 128) f32

    dim3 grid(BATCH);
    dim3 block(32);
    embbag_wmma_kernel<<<grid, block, 0, stream>>>(x, w, table, out);
}
